// MultiChannelMultiAttentionCNN_18494129176681
// MI455X (gfx1250) — matrix operand order repaired and checked by reading, not compiled, not run on hardware
//
#include <hip/hip_runtime.h>
#include <math.h>

typedef __bf16 bf16_t;
typedef __attribute__((ext_vector_type(16))) __bf16 v16bf;
typedef __attribute__((ext_vector_type(8)))  float  v8f;

#define S_LEN 1024
#define E_DIM 64
#define B_SZ  2
#define N_ROWS (B_SZ * S_LEN)   // 2048
#define EPS 1e-5f

#if defined(__has_builtin)
#if __has_builtin(__builtin_amdgcn_global_load_async_to_lds_b32)
#define HAVE_ASYNC_LDS 1
#endif
#if __has_builtin(__builtin_amdgcn_s_wait_asynccnt)
#define HAVE_WAIT_ASYNC 1
#endif
#endif

// ---- WMMA fragment index helpers (wave32, 16x16x32 bf16) ----
// A matrix 16x32: lane = {half = lane>>4, m = lane&15}; element e (0..15):
//   VGPR r=e>>1 pair p=e&1 ; K = (r>>2)*16 + half*8 + (r&3)*2 + p
__device__ __forceinline__ int ka_of(int e, int half) {
  int r = e >> 1, p = e & 1;
  return ((r >> 2) << 4) + (half << 3) + ((r & 3) << 1) + p;
}
// B matrix 32x16: lane = {half, n = lane&15}; K = (r>>2)*16 + half*8 + (r&3)*2 + p (same as A)
__device__ __forceinline__ int kb_of(int e, int half) {
  int r = e >> 1, p = e & 1;
  return ((r >> 2) << 4) + (half << 3) + ((r & 3) << 1) + p;
}
// C/D 16x16 f32: element r (0..7): m = r + 8*half, n = lane&15

// ---- async global->LDS helpers (ASYNCcnt path on gfx1250) ----
__device__ __forceinline__ void lds_async_copy4(const bf16_t* g, bf16_t* l) {
#if defined(HAVE_ASYNC_LDS)
  __builtin_amdgcn_global_load_async_to_lds_b32((int*)g, (int*)l, 0, 0);
#else
  *(uint32_t*)l = *(const uint32_t*)g;
#endif
}
__device__ __forceinline__ void lds_async_wait_le1() {
#if defined(HAVE_ASYNC_LDS)
#if defined(HAVE_WAIT_ASYNC)
  __builtin_amdgcn_s_wait_asynccnt(1);
#else
  asm volatile("s_wait_asynccnt 0x1" ::: "memory");
#endif
#endif
}
__device__ __forceinline__ void lds_async_wait_le0() {
#if defined(HAVE_ASYNC_LDS)
#if defined(HAVE_WAIT_ASYNC)
  __builtin_amdgcn_s_wait_asynccnt(0);
#else
  asm volatile("s_wait_asynccnt 0x0" ::: "memory");
#endif
#endif
}

// ============================================================
__global__ void k_cast_bf16(const float* __restrict__ s, bf16_t* __restrict__ d, int n) {
  int i = blockIdx.x * blockDim.x + threadIdx.x;
  if (i < n) d[i] = (bf16_t)s[i];
}

// q = x Wq^T ; k = x Wk^T + bk ; v = x Wv^T   (all [2048,64], bf16 out)
__global__ void k_proj(const bf16_t* __restrict__ xb,
                       const float* __restrict__ Wq, const float* __restrict__ Wk,
                       const float* __restrict__ bk, const float* __restrict__ Wv,
                       bf16_t* __restrict__ qo, bf16_t* __restrict__ ko, bf16_t* __restrict__ vo) {
  int lane = threadIdx.x, half = lane >> 4, ln = lane & 15;
  int m0 = blockIdx.x * 16, n0 = blockIdx.y * 16;
  v8f accq = {}, acck = {}, accv = {};
  for (int kb = 0; kb < E_DIM; kb += 32) {
    v16bf a, bq, bk2, bv;
#pragma unroll
    for (int e = 0; e < 16; ++e) {
      a[e] = xb[(size_t)(m0 + ln) * E_DIM + kb + ka_of(e, half)];
      int kB = kb + kb_of(e, half);
      bq[e]  = (bf16_t)Wq[(size_t)(n0 + ln) * E_DIM + kB];
      bk2[e] = (bf16_t)Wk[(size_t)(n0 + ln) * E_DIM + kB];
      bv[e]  = (bf16_t)Wv[(size_t)(n0 + ln) * E_DIM + kB];
    }
    accq = __builtin_amdgcn_wmma_f32_16x16x32_bf16(false, a, false, bq,  (short)0, accq, false, false);
    acck = __builtin_amdgcn_wmma_f32_16x16x32_bf16(false, a, false, bk2, (short)0, acck, false, false);
    accv = __builtin_amdgcn_wmma_f32_16x16x32_bf16(false, a, false, bv,  (short)0, accv, false, false);
  }
#pragma unroll
  for (int r = 0; r < 8; ++r) {
    int m = m0 + r + 8 * half, n = n0 + ln;
    qo[(size_t)m * E_DIM + n] = (bf16_t)accq[r];
    ko[(size_t)m * E_DIM + n] = (bf16_t)(acck[r] + bk[n]);
    vo[(size_t)m * E_DIM + n] = (bf16_t)accv[r];
  }
}

// scores[b,q,k] = <A[b,q,:], B[b,k,:]> * scale, causal-masked to -1e9
__global__ void k_scores(const bf16_t* __restrict__ A, const bf16_t* __restrict__ Bm,
                         float* __restrict__ out, const float* __restrict__ scale_p,
                         int use_scale) {
  int lane = threadIdx.x, half = lane >> 4, ln = lane & 15;
  int k0 = blockIdx.x * 16, q0 = blockIdx.y * 16, b = blockIdx.z;
  float* o = out + ((size_t)b * S_LEN + q0) * S_LEN;
  if (k0 > q0 + 15) {            // tile entirely above diagonal: skip WMMA
#pragma unroll
    for (int r = 0; r < 8; ++r)
      o[(size_t)(r + 8 * half) * S_LEN + k0 + ln] = -1e9f;
    return;
  }
  const bf16_t* Ab = A + (size_t)b * S_LEN * E_DIM;
  const bf16_t* Bb = Bm + (size_t)b * S_LEN * E_DIM;
  v8f acc = {};
  for (int kb = 0; kb < E_DIM; kb += 32) {
    v16bf a, bb;
#pragma unroll
    for (int e = 0; e < 16; ++e) {
      a[e]  = Ab[(size_t)(q0 + ln) * E_DIM + kb + ka_of(e, half)];
      bb[e] = Bb[(size_t)(k0 + ln) * E_DIM + kb + kb_of(e, half)];
    }
    acc = __builtin_amdgcn_wmma_f32_16x16x32_bf16(false, a, false, bb, (short)0, acc, false, false);
  }
  float sc = use_scale ? (0.125f * scale_p[0]) : 1.0f;  // 1/sqrt(64) * scale
#pragma unroll
  for (int r = 0; r < 8; ++r) {
    int q = q0 + r + 8 * half, k = k0 + ln;
    float v = acc[r] * sc;
    if (k > q) v = -1e9f;
    o[(size_t)(r + 8 * half) * S_LEN + k] = v;
  }
}

// s3[b,q,k] = sum_d tanh(x[b,q,d] + x[b,k,d]), masked
__global__ void k_tanh_scores(const float* __restrict__ x, float* __restrict__ out) {
  __shared__ float xq[E_DIM];
  int b = blockIdx.z, q = blockIdx.y;
  int k = blockIdx.x * blockDim.x + threadIdx.x;
  if (threadIdx.x < E_DIM)
    xq[threadIdx.x] = x[((size_t)b * S_LEN + q) * E_DIM + threadIdx.x];
  __syncthreads();
  float s;
  if (k <= q) {
    const float* xk = x + ((size_t)b * S_LEN + k) * E_DIM;
    s = 0.f;
#pragma unroll 8
    for (int d = 0; d < E_DIM; ++d) s += tanhf(xq[d] + xk[d]);
  } else {
    s = -1e9f;
  }
  out[((size_t)b * S_LEN + q) * S_LEN + k] = s;
}

// row softmax over 1024, output bf16 probabilities
__global__ void k_softmax_p(const float* __restrict__ scores, bf16_t* __restrict__ P) {
  __shared__ float red[256];
  size_t row = blockIdx.x;
  const float* r = scores + row * S_LEN;
  bf16_t* p = P + row * S_LEN;
  int t = threadIdx.x;
  float mx = -1e30f;
#pragma unroll
  for (int i = 0; i < 4; ++i) mx = fmaxf(mx, r[t + 256 * i]);
  red[t] = mx; __syncthreads();
  for (int s = 128; s > 0; s >>= 1) { if (t < s) red[t] = fmaxf(red[t], red[t + s]); __syncthreads(); }
  mx = red[0]; __syncthreads();
  float e0[4], sum = 0.f;
#pragma unroll
  for (int i = 0; i < 4; ++i) { e0[i] = __expf(r[t + 256 * i] - mx); sum += e0[i]; }
  red[t] = sum; __syncthreads();
  for (int s = 128; s > 0; s >>= 1) { if (t < s) red[t] += red[t + s]; __syncthreads(); }
  float inv = 1.0f / red[0];
#pragma unroll
  for (int i = 0; i < 4; ++i) p[t + 256 * i] = (bf16_t)(e0[i] * inv);
}

// att = sum_br w_br * (P_br @ V_br);  V = {x, v, x}
__global__ void k_pv(const bf16_t* __restrict__ P, const bf16_t* __restrict__ xb,
                     const bf16_t* __restrict__ vb, const float* __restrict__ aw,
                     float* __restrict__ att) {
  int lane = threadIdx.x, half = lane >> 4, ln = lane & 15;
  int n0 = blockIdx.x * 16, m0 = blockIdx.y * 16, b = blockIdx.z;
  float wsum = aw[0] + aw[1] + aw[2];
  v8f tot = {};
  for (int br = 0; br < 3; ++br) {
    const bf16_t* Pb = P + (((size_t)br * B_SZ + b) * S_LEN) * S_LEN;
    const bf16_t* V  = (br == 1 ? vb : xb) + (size_t)b * S_LEN * E_DIM;
    v8f acc = {};
    for (int kb = 0; kb < S_LEN; kb += 32) {
      v16bf a, bb;
#pragma unroll
      for (int e = 0; e < 16; ++e) {
        a[e]  = Pb[(size_t)(m0 + ln) * S_LEN + kb + ka_of(e, half)];
        bb[e] = V[(size_t)(kb + kb_of(e, half)) * E_DIM + n0 + ln];
      }
      acc = __builtin_amdgcn_wmma_f32_16x16x32_bf16(false, a, false, bb, (short)0, acc, false, false);
    }
    float w = aw[br] / wsum;
#pragma unroll
    for (int r = 0; r < 8; ++r) tot[r] += w * acc[r];
  }
#pragma unroll
  for (int r = 0; r < 8; ++r)
    att[((size_t)b * S_LEN + m0 + r + 8 * half) * E_DIM + n0 + ln] = tot[r];
}

// batch-norm stats (over B*S) for x, x+att, att; one block per feature e
__global__ void k_prebn_stats(const float* __restrict__ x, const float* __restrict__ att,
                              float* __restrict__ st) {
  int e = blockIdx.x, t = threadIdx.x;
  float sx = 0, sxx = 0, sa = 0, saa = 0, sxa = 0;
  for (int s = t; s < N_ROWS; s += 256) {
    float xv = x[(size_t)s * E_DIM + e];
    float av = att[(size_t)s * E_DIM + e];
    sx += xv; sxx += xv * xv; sa += av; saa += av * av; sxa += xv * av;
  }
  __shared__ float r0[256], r1[256], r2[256], r3[256], r4[256];
  r0[t] = sx; r1[t] = sxx; r2[t] = sa; r3[t] = saa; r4[t] = sxa;
  __syncthreads();
  for (int s = 128; s > 0; s >>= 1) {
    if (t < s) { r0[t]+=r0[t+s]; r1[t]+=r1[t+s]; r2[t]+=r2[t+s]; r3[t]+=r3[t+s]; r4[t]+=r4[t+s]; }
    __syncthreads();
  }
  if (t == 0) {
    float inv = 1.0f / (float)N_ROWS;
    float mx = r0[0] * inv, ma = r2[0] * inv;
    float vx = r1[0] * inv - mx * mx;
    float va = r3[0] * inv - ma * ma;
    float mm = mx + ma;
    float vm = (r1[0] + 2.f * r4[0] + r3[0]) * inv - mm * mm;
    float* o = st + e * 6;
    o[0] = mx; o[1] = vx; o[2] = mm; o[3] = vm; o[4] = ma; o[5] = va;
  }
}

__global__ void k_prebn_apply(const float* __restrict__ x, const float* __restrict__ att,
                              const float* __restrict__ st,
                              const float* __restrict__ g1, const float* __restrict__ b1,
                              const float* __restrict__ g2, const float* __restrict__ b2,
                              const float* __restrict__ g3, const float* __restrict__ b3,
                              bf16_t* __restrict__ l1, bf16_t* __restrict__ l2,
                              bf16_t* __restrict__ l3) {
  int i = blockIdx.x * blockDim.x + threadIdx.x;
  if (i >= N_ROWS * E_DIM) return;
  int e = i % E_DIM;
  const float* s = st + e * 6;
  float xv = x[i], av = att[i], mv = xv + av;
  l1[i] = (bf16_t)((xv - s[0]) * rsqrtf(s[1] + EPS) * g1[e] + b1[e]);
  l2[i] = (bf16_t)((mv - s[2]) * rsqrtf(s[3] + EPS) * g2[e] + b2[e]);
  l3[i] = (bf16_t)((av - s[4]) * rsqrtf(s[5] + EPS) * g3[e] + b3[e]);
}

// Prepack conv weights fp32 -> bf16 in WMMA fragment order:
//   Wp[(((mt*(K/32)+ks)*32 + lane)*16 + e] = W[mt*16 + (lane&15)][ks*32 + ka_of(e,lane>>4)]
// so each lane's A fragment for slab ks is one contiguous 32-byte run.
__global__ void k_prepack_w(const float* __restrict__ W, bf16_t* __restrict__ Wp, int K) {
  int idx = blockIdx.x * blockDim.x + threadIdx.x;
  if (idx >= S_LEN * K) return;
  int e    = idx & 15;
  int lane = (idx >> 4) & 31;
  int ks   = (idx >> 9) % (K >> 5);
  int mt   = (idx >> 9) / (K >> 5);
  int half = lane >> 4, m = lane & 15;
  const float* wp = W + (size_t)(mt * 16 + m) * K + ks * 32 + ka_of(e, half);
  __builtin_prefetch(wp + 256, 0, 0);   // global_prefetch_b8 on the weight stream
  Wp[idx] = (bf16_t)wp[0];
}

// Conv1d(in_ch=out_ch=1024, kernel=KSZ) as GEMM. One M-tile (16 out channels)
// per block; 8 waves cover the full N = B*Lout range. Prepacked bf16 weight
// slabs (512 elem = 1 KB) are double-buffered into LDS via async global->LDS
// copies (ASYNCcnt), each lane then reads its A fragment as one 32 B LDS load.
// Fused bias + ReLU. y layout: [B][1024][Lout] fp32.
template <int KSZ>
__global__ void k_conv_gemm(const bf16_t* __restrict__ Wp, const float* __restrict__ bias,
                            const bf16_t* __restrict__ src, float* __restrict__ y,
                            int Lin) {
  __shared__ alignas(32) bf16_t Abuf[2][512];
  const int Lout = Lin - KSZ + 1;
  const int N = B_SZ * Lout;
  const int K = S_LEN * KSZ;
  const int nslab = K >> 5;
  int tid = threadIdx.x;
  int lane = tid & 31, half = lane >> 4, ln = lane & 15;
  int wave = tid >> 5;
  int n = wave * 16 + ln;
  int bIdx = n / Lout, l = n % Lout;        // only meaningful when n < N
  const bf16_t* Wbase = Wp + (size_t)blockIdx.x * K * 16;  // this M-tile's slabs
  const int sliceOff = wave * 64 + lane * 2;               // 4 B per lane per slab

  lds_async_copy4(Wbase + sliceOff, &Abuf[0][sliceOff]);   // prologue: slab 0
  v8f acc = {};
  for (int i = 0; i < nslab; ++i) {
    int cur = i & 1;
    if (i + 1 < nslab) {
      lds_async_copy4(Wbase + (size_t)(i + 1) * 512 + sliceOff, &Abuf[cur ^ 1][sliceOff]);
      lds_async_wait_le1();                // my slab-i slice landed (in-order)
    } else {
      lds_async_wait_le0();
    }
    __syncthreads();                       // all waves' slices of slab i landed
    v16bf a = *(const v16bf*)&Abuf[cur][lane * 16];  // 2x ds_load_b128
    v16bf bb;
#pragma unroll
    for (int e = 0; e < 16; ++e) {
      int kcol = (i << 5) + kb_of(e, half);
      int ch = kcol / KSZ, kc = kcol - ch * KSZ;
      bb[e] = (n < N) ? src[((size_t)bIdx * S_LEN + ch) * Lin + l + kc] : (bf16_t)0.f;
    }
    acc = __builtin_amdgcn_wmma_f32_16x16x32_bf16(false, a, false, bb, (short)0, acc, false, false);
    __syncthreads();                       // reads done before slab i+2 overwrites
  }
  if (n < N) {
#pragma unroll
    for (int r = 0; r < 8; ++r) {
      int o = blockIdx.x * 16 + r + 8 * half;
      y[((size_t)bIdx * S_LEN + o) * Lout + l] = fmaxf(acc[r] + bias[o], 0.f);
    }
  }
}

__global__ void k_postbn_stats(const float* __restrict__ y, float* __restrict__ st, int Lout) {
  int f = blockIdx.x, t = threadIdx.x;
  float s1 = 0, s2 = 0;
  for (int s = t; s < N_ROWS; s += 256) {
    float v = y[(size_t)s * Lout + f];
    s1 += v; s2 += v * v;
  }
  __shared__ float r0[256], r1[256];
  r0[t] = s1; r1[t] = s2; __syncthreads();
  for (int s = 128; s > 0; s >>= 1) {
    if (t < s) { r0[t] += r0[t + s]; r1[t] += r1[t + s]; }
    __syncthreads();
  }
  if (t == 0) {
    float inv = 1.0f / (float)N_ROWS;
    float m = r0[0] * inv;
    st[f * 2] = m; st[f * 2 + 1] = r1[0] * inv - m * m;
  }
}

__global__ void k_postbn_apply(const float* __restrict__ y, const float* __restrict__ st,
                               const float* __restrict__ g, const float* __restrict__ b,
                               bf16_t* __restrict__ dbf, float* __restrict__ df,
                               int Lout, int write_f) {
  int i = blockIdx.x * blockDim.x + threadIdx.x;
  if (i >= N_ROWS * Lout) return;
  int f = i % Lout;
  float v = (y[i] - st[f * 2]) * rsqrtf(st[f * 2 + 1] + EPS) * g[f] + b[f];
  dbf[i] = (bf16_t)v;
  if (write_f) df[i] = v;
}

struct FinalArgs {
  const float* seg[9];   // [inception][segment 0:64 / 1:62 / 2:58]
  const float* sw; const float* lng; const float* lnb;
  float* out;
};

__global__ void k_final(FinalArgs fa) {
  __shared__ float r0[256], r1[256];
  int s = blockIdx.x, t = threadIdx.x;
  float wsum = fa.sw[0] + fa.sw[1] + fa.sw[2];
  bool act = t < 184;
  int sg = 0, off = 0, Lw = 64;
  if (t >= 126)      { sg = 2; off = t - 126; Lw = 58; }
  else if (t >= 64)  { sg = 1; off = t - 64;  Lw = 62; }
  else               { sg = 0; off = t;       Lw = 64; }
  float val = 0.f;
  if (act)
    for (int inc = 0; inc < 3; ++inc)
      val += (fa.sw[inc] / wsum) * fa.seg[inc * 3 + sg][(size_t)s * Lw + off];
  r0[t] = act ? val : 0.f;
  r1[t] = act ? val * val : 0.f;
  __syncthreads();
  for (int k = 128; k > 0; k >>= 1) {
    if (t < k) { r0[t] += r0[t + k]; r1[t] += r1[t + k]; }
    __syncthreads();
  }
  float mean = r0[0] * (1.0f / 184.f);
  float var  = r1[0] * (1.0f / 184.f) - mean * mean;
  if (act)
    fa.out[(size_t)s * 184 + t] = (val - mean) * rsqrtf(var + EPS) * fa.lng[t] + fa.lnb[t];
}

// ============================================================
extern "C" void kernel_launch(void* const* d_in, const int* in_sizes, int n_in,
                              void* d_out, int out_size, void* d_ws, size_t ws_size,
                              hipStream_t stream) {
  (void)in_sizes; (void)n_in; (void)out_size; (void)ws_size;
  const float* x     = (const float*)d_in[0];
  const float* aw    = (const float*)d_in[1];
  const float* scale = (const float*)d_in[2];
  const float* Wq    = (const float*)d_in[3];
  const float* Wk    = (const float*)d_in[4];
  const float* bkv   = (const float*)d_in[5];
  const float* Wv    = (const float*)d_in[6];
  const float* lng   = (const float*)d_in[85];
  const float* lnb   = (const float*)d_in[86];
  const float* sw    = (const float*)d_in[87];
  float* out = (float*)d_out;

  char* ws = (char*)d_ws;
  size_t off = 0;
  auto take = [&](size_t bytes) -> char* {
    char* p = ws + off;
    off = (off + bytes + 255) & ~(size_t)255;
    return p;
  };

  bf16_t* x_bf = (bf16_t*)take((size_t)N_ROWS * E_DIM * 2);
  bf16_t* q_bf = (bf16_t*)take((size_t)N_ROWS * E_DIM * 2);
  bf16_t* k_bf = (bf16_t*)take((size_t)N_ROWS * E_DIM * 2);
  bf16_t* v_bf = (bf16_t*)take((size_t)N_ROWS * E_DIM * 2);
  float*  scores = (float*)take(3ull * B_SZ * S_LEN * S_LEN * 4);   // 24 MB
  bf16_t* P      = (bf16_t*)take(3ull * B_SZ * S_LEN * S_LEN * 2);  // 12 MB
  float*  att    = (float*)take((size_t)N_ROWS * E_DIM * 4);
  float*  st_pre = (float*)take(64 * 6 * 4);
  bf16_t* l123[3];
  for (int i = 0; i < 3; ++i) l123[i] = (bf16_t*)take((size_t)N_ROWS * E_DIM * 2);
  float*  ybuf   = (float*)take((size_t)N_ROWS * E_DIM * 4);
  bf16_t* t1     = (bf16_t*)take((size_t)N_ROWS * E_DIM * 2);
  bf16_t* t2     = (bf16_t*)take((size_t)N_ROWS * E_DIM * 2);
  float*  st_post = (float*)take(64 * 2 * 4);
  bf16_t* wpack  = (bf16_t*)take((size_t)S_LEN * S_LEN * 5 * 2);    // 10.5 MB, reused per conv
  float*  term[9];
  for (int inc = 0; inc < 3; ++inc) {
    term[inc * 3 + 0] = (float*)take((size_t)N_ROWS * 64 * 4);
    term[inc * 3 + 1] = (float*)take((size_t)N_ROWS * 62 * 4);
    term[inc * 3 + 2] = (float*)take((size_t)N_ROWS * 58 * 4);
  }

  const int nx = N_ROWS * E_DIM;  // 131072
  k_cast_bf16<<<(nx + 255) / 256, 256, 0, stream>>>(x, x_bf, nx);
  k_proj<<<dim3(N_ROWS / 16, E_DIM / 16), 32, 0, stream>>>(x_bf, Wq, Wk, bkv, Wv, q_bf, k_bf, v_bf);
  k_scores<<<dim3(64, 64, B_SZ), 32, 0, stream>>>(x_bf, x_bf, scores, scale, 0);
  k_scores<<<dim3(64, 64, B_SZ), 32, 0, stream>>>(q_bf, k_bf,
      scores + (size_t)B_SZ * S_LEN * S_LEN, scale, 1);
  k_tanh_scores<<<dim3(4, S_LEN, B_SZ), 256, 0, stream>>>(x,
      scores + 2ull * B_SZ * S_LEN * S_LEN);
  k_softmax_p<<<3 * B_SZ * S_LEN, 256, 0, stream>>>(scores, P);
  k_pv<<<dim3(E_DIM / 16, 64, B_SZ), 32, 0, stream>>>(P, x_bf, v_bf, aw, att);
  k_prebn_stats<<<64, 256, 0, stream>>>(x, att, st_pre);
  k_prebn_apply<<<(nx + 255) / 256, 256, 0, stream>>>(x, att, st_pre,
      (const float*)d_in[7],  (const float*)d_in[8],
      (const float*)d_in[9],  (const float*)d_in[10],
      (const float*)d_in[11], (const float*)d_in[12],
      l123[0], l123[1], l123[2]);

  auto conv_bn = [&](const float* W, const float* bias, const bf16_t* srcp,
                     int Lin, int ksz, const float* g, const float* bparam,
                     bf16_t* dbf, float* df, int wf) {
    int Lout = Lin - ksz + 1;
    int K = S_LEN * ksz;
    int npk = S_LEN * K;
    k_prepack_w<<<(npk + 255) / 256, 256, 0, stream>>>(W, wpack, K);
    switch (ksz) {
      case 1: k_conv_gemm<1><<<64, 256, 0, stream>>>(wpack, bias, srcp, ybuf, Lin); break;
      case 3: k_conv_gemm<3><<<64, 256, 0, stream>>>(wpack, bias, srcp, ybuf, Lin); break;
      default: k_conv_gemm<5><<<64, 256, 0, stream>>>(wpack, bias, srcp, ybuf, Lin); break;
    }
    k_postbn_stats<<<Lout, 256, 0, stream>>>(ybuf, st_post, Lout);
    int n = N_ROWS * Lout;
    k_postbn_apply<<<(n + 255) / 256, 256, 0, stream>>>(ybuf, st_post, g, bparam,
                                                        dbf, df, Lout, wf);
  };

  for (int inc = 0; inc < 3; ++inc) {
    int base = 13 + 24 * inc;
    auto PP = [&](int i) { return (const float*)d_in[base + i]; };
    bf16_t* src = l123[inc];
    // branch 1: c11(k=1) -> bn11
    conv_bn(PP(0), PP(1), src, 64, 1, PP(12), PP(13), t1, term[inc * 3 + 0], 1);
    // branch 2: c21(k=1)->bn21 ; c22(k=3)->bn22
    conv_bn(PP(2), PP(3), src, 64, 1, PP(14), PP(15), t1, term[inc * 3 + 0], 0);
    conv_bn(PP(4), PP(5), t1, 64, 3, PP(16), PP(17), t2, term[inc * 3 + 1], 1);
    // branch 3: c31(k=1)->bn31 ; c32(k=3)->bn32 ; c33(k=5)->bn33
    conv_bn(PP(6), PP(7), src, 64, 1, PP(18), PP(19), t1, term[inc * 3 + 0], 0);
    conv_bn(PP(8), PP(9), t1, 64, 3, PP(20), PP(21), t2, term[inc * 3 + 1], 0);
    conv_bn(PP(10), PP(11), t2, 62, 5, PP(22), PP(23), t1, term[inc * 3 + 2], 1);
  }

  FinalArgs fa;
  for (int i = 0; i < 9; ++i) fa.seg[i] = term[i];
  fa.sw = sw; fa.lng = lng; fa.lnb = lnb; fa.out = out;
  k_final<<<N_ROWS, 256, 0, stream>>>(fa);
}
